// TwoLayerGCN_29953101922492
// MI455X (gfx1250) — compile-verified, matrix operand order checked by reading, not run on hardware
//
#include <hip/hip_runtime.h>
#include <hip/hip_bf16.h>

#define NN 100000
#define NE 3200000

typedef __attribute__((ext_vector_type(2)))  float  v2f;
typedef __attribute__((ext_vector_type(8)))  float  v8f;
typedef __attribute__((ext_vector_type(16))) __bf16 v16bf;

// ---------------- degree / norm ----------------
__global__ void k_deg_init(float* deg) {
    int i = blockIdx.x * blockDim.x + threadIdx.x;
    if (i < NN) deg[i] = 1.0f;               // self-loop weight
}

__global__ void k_deg_accum(const int* __restrict__ col,
                            const float* __restrict__ ew,
                            float* __restrict__ deg) {
    int e = blockIdx.x * blockDim.x + threadIdx.x;
    if (e < NE) atomicAdd(&deg[col[e]], ew[e]);
}

__global__ void k_rsqrt_inplace(float* deg) {
    int i = blockIdx.x * blockDim.x + threadIdx.x;
    if (i < NN) {
        float d = deg[i];
        deg[i] = (d > 0.0f) ? rsqrtf(d) : 0.0f;
    }
}

// ---------------- WMMA GEMM:  H[M x N] = X[M x K] @ W[K x N] ----------------
// One wave computes one 16-row M-tile across ALL N/16 column tiles (A reuse).
template <int K, int N>
__global__ __launch_bounds__(128) void k_gemm_wmma(const float* __restrict__ X,
                                                   const float* __restrict__ W,
                                                   float* __restrict__ H) {
    constexpr int NT = N / 16;
    const int wave = blockIdx.x * 4 + (threadIdx.x >> 5);
    if (wave >= NN / 16) return;             // wave-uniform exit (EXEC all-1 for WMMA)
    const int lane = threadIdx.x & 31;
    const int half = lane >> 4;              // selects K sub-pair / M sub-block
    const int l16  = lane & 15;
    const int m0   = wave * 16;

    v8f acc[NT];
#pragma unroll
    for (int t = 0; t < NT; ++t) acc[t] = (v8f){0,0,0,0,0,0,0,0};

    const float* xrow = X + (size_t)(m0 + l16) * K;

#if __has_builtin(__builtin_amdgcn_wmma_f32_16x16x4_f32)
    // fp32 path: A 16x4 per step, lane holds (K=2*half, K=2*half+1) of row l16.
#pragma unroll
    for (int kk = 0; kk < K; kk += 4) {
        v2f a = *(const v2f*)(xrow + kk + 2 * half);
#pragma unroll
        for (int t = 0; t < NT; ++t) {
            v2f b;
            b.x = W[(size_t)(kk + 2 * half)     * N + t * 16 + l16];
            b.y = W[(size_t)(kk + 2 * half + 1) * N + t * 16 + l16];
            acc[t] = __builtin_amdgcn_wmma_f32_16x16x4_f32(
                false, a, false, b, (short)0, acc[t], false, false);
        }
    }
#else
    // bf16 fallback: 16x16x32, documented 16-bit A/B striping.
#pragma unroll
    for (int kk = 0; kk < K; kk += 32) {
        v16bf a;
#pragma unroll
        for (int v = 0; v < 8; ++v) {
            int kbase = kk + ((v < 4) ? 0 : 16) + 8 * half + 2 * (v & 3);
            a[2 * v]     = (__bf16)xrow[kbase];
            a[2 * v + 1] = (__bf16)xrow[kbase + 1];
        }
#pragma unroll
        for (int t = 0; t < NT; ++t) {
            v16bf b;
            int n = t * 16 + l16;
#pragma unroll
            for (int v = 0; v < 8; ++v) {
                int kbase = kk + ((v < 4) ? 0 : 16) + 8 * half + 2 * (v & 3);
                b[2 * v]     = (__bf16)W[(size_t)kbase * N + n];
                b[2 * v + 1] = (__bf16)W[(size_t)(kbase + 1) * N + n];
            }
            acc[t] = __builtin_amdgcn_wmma_f32_16x16x32_bf16(
                false, a, false, b, (short)0, acc[t], false, false);
        }
    }
#endif

    // C/D layout: vgpr j -> M = 8*half + j ; lane&15 -> N column.
#pragma unroll
    for (int t = 0; t < NT; ++t)
#pragma unroll
        for (int j = 0; j < 8; ++j)
            H[(size_t)(m0 + 8 * half + j) * N + t * 16 + l16] = acc[t][j];
}

// ---------------- aggregation ----------------
// out[i,f] = dinv[i]^2 * H[i,f]   (self-loop term; also initializes accumulator)
template <int D>
__global__ void k_self_init(const float* __restrict__ dinv,
                            const float* __restrict__ H,
                            float* __restrict__ out) {
    unsigned t = blockIdx.x * blockDim.x + threadIdx.x;
    if (t >= (unsigned)NN * D) return;
    unsigned i = t / D;
    float s = dinv[i];
    out[t] = s * s * H[t];
}

// out[c,f] += dinv[r]*ew*dinv[c] * H[r,f]   (lane-per-feature, L2-resident)
template <int D>
__global__ void k_scatter(const int* __restrict__ row,
                          const int* __restrict__ col,
                          const float* __restrict__ ew,
                          const float* __restrict__ dinv,
                          const float* __restrict__ H,
                          float* __restrict__ out) {
    unsigned t = blockIdx.x * blockDim.x + threadIdx.x;
    if (t >= (unsigned)NE * D) return;
    unsigned e = t / D;
    unsigned f = t & (D - 1);
    int r = row[e], c = col[e];
    if (f == 0) {
        unsigned e2 = e + 4096;
        if (e2 < NE) __builtin_prefetch(&H[(size_t)row[e2] * D], 0, 1);
    }
    float nrm = dinv[r] * ew[e] * dinv[c];
    atomicAdd(&out[(size_t)c * D + f], nrm * H[(size_t)r * D + f]);
}

template <int D, bool RELU>
__global__ void k_bias_act(float* __restrict__ out, const float* __restrict__ bias) {
    unsigned t = blockIdx.x * blockDim.x + threadIdx.x;
    if (t >= (unsigned)NN * D) return;
    float v = out[t] + bias[t & (D - 1)];
    out[t] = RELU ? fmaxf(v, 0.0f) : v;
}

// ---------------- launcher ----------------
extern "C" void kernel_launch(void* const* d_in, const int* in_sizes, int n_in,
                              void* d_out, int out_size, void* d_ws, size_t ws_size,
                              hipStream_t stream) {
    const float* x   = (const float*)d_in[0];
    const int*   ei  = (const int*)d_in[1];
    const float* ew  = (const float*)d_in[2];
    const float* W1  = (const float*)d_in[3];
    const float* b1  = (const float*)d_in[4];
    const float* W2  = (const float*)d_in[5];
    const float* b2  = (const float*)d_in[6];
    const int* row = ei;
    const int* col = ei + NE;

    float* ws   = (float*)d_ws;
    float* dinv = ws;                                  // NN
    float* H1   = dinv + ((NN + 63) & ~63);            // NN*64
    float* A1   = H1 + (size_t)NN * 64;                // NN*64
    float* H2   = A1 + (size_t)NN * 64;                // NN*32
    float* O    = (float*)d_out;                       // NN*32

    const int B = 256;

    // normalization
    k_deg_init<<<(NN + B - 1) / B, B, 0, stream>>>(dinv);
    k_deg_accum<<<(NE + B - 1) / B, B, 0, stream>>>(col, ew, dinv);
    k_rsqrt_inplace<<<(NN + B - 1) / B, B, 0, stream>>>(dinv);

    const int tiles = NN / 16;                         // 6250, exact
    const int gblk  = (tiles + 3) / 4;                 // 4 waves / block

    // ---- layer 1 ----
    k_gemm_wmma<64, 64><<<gblk, 128, 0, stream>>>(x, W1, H1);
    k_self_init<64><<<((unsigned)NN * 64 + B - 1) / B, B, 0, stream>>>(dinv, H1, A1);
    k_scatter<64><<<((unsigned)NE * 64 + B - 1) / B, B, 0, stream>>>(row, col, ew, dinv, H1, A1);
    k_bias_act<64, true><<<((unsigned)NN * 64 + B - 1) / B, B, 0, stream>>>(A1, b1);

    // ---- layer 2 ----
    k_gemm_wmma<64, 32><<<gblk, 128, 0, stream>>>(A1, W2, H2);
    k_self_init<32><<<((unsigned)NN * 32 + B - 1) / B, B, 0, stream>>>(dinv, H2, O);
    k_scatter<32><<<((unsigned)NE * 32 + B - 1) / B, B, 0, stream>>>(row, col, ew, dinv, H2, O);
    k_bias_act<32, false><<<((unsigned)NN * 32 + B - 1) / B, B, 0, stream>>>(O, b2);
}
